// RNNWWD_8873402433903
// MI455X (gfx1250) — compile-verified, matrix operand order checked by reading, not run on hardware
//
#include <hip/hip_runtime.h>
#include <hip/hip_bf16.h>
#include <math.h>

// ---------------- problem constants ----------------
#define NODE    800
#define BATCH   80      // number of TR windows (outer scan)
#define HIDDEN  150     // euler steps per TR
#define NWG     25      // persistent workgroups
#define RPW     32      // rows (nodes) per workgroup
#define KBLK    (NODE / 32)   // 25 K-chunks of 32 for wmma 16x16x32

typedef __bf16 bf16_t;
typedef __attribute__((ext_vector_type(8)))  __bf16 v8bf;
typedef __attribute__((ext_vector_type(16))) __bf16 v16bf;
typedef __attribute__((ext_vector_type(8)))  float  v8f;

// ---------------- prep kernel 1: ws = 0.5*(w + w^T), per-row sum-of-squares ----------------
__global__ void __launch_bounds__(256)
prep1_kernel(const float* __restrict__ wbb, const float* __restrict__ sc,
             float* __restrict__ wsmat, float* __restrict__ partials) {
    __shared__ float red[256];
    const int i = blockIdx.x, tid = threadIdx.x;
    float ss = 0.f;
    for (int j = tid; j < NODE; j += 256) {
        float wij = (1.f + tanhf(wbb[i * NODE + j])) * sc[i * NODE + j];
        float wji = (1.f + tanhf(wbb[j * NODE + i])) * sc[j * NODE + i];
        float val = 0.5f * (wij + wji);
        wsmat[i * NODE + j] = val;
        ss += val * val;
    }
    red[tid] = ss; __syncthreads();
    for (int s = 128; s > 0; s >>= 1) {
        if (tid < s) red[tid] += red[tid + s];
        __syncthreads();
    }
    if (tid == 0) partials[i] = red[0];
}

// ---------------- prep kernel 2: l_s = ws/||ws|| - diag(rowsum), emitted as bf16 ----------------
__global__ void __launch_bounds__(256)
prep2_kernel(const float* __restrict__ wsmat, const float* __restrict__ partials,
             bf16_t* __restrict__ lsbf) {
    __shared__ float red[256];
    __shared__ float sNinv;
    const int i = blockIdx.x, tid = threadIdx.x;

    float t = 0.f;
    for (int j = tid; j < NODE; j += 256) t += partials[j];
    red[tid] = t; __syncthreads();
    for (int s = 128; s > 0; s >>= 1) {
        if (tid < s) red[tid] += red[tid + s];
        __syncthreads();
    }
    if (tid == 0) sNinv = 1.0f / sqrtf(red[0]);
    __syncthreads();
    const float ninv = sNinv;

    float rs = 0.f;
    for (int j = tid; j < NODE; j += 256) rs += wsmat[i * NODE + j];
    red[tid] = rs; __syncthreads();
    for (int s = 128; s > 0; s >>= 1) {
        if (tid < s) red[tid] += red[tid + s];
        __syncthreads();
    }
    const float rowsum = red[0];

    for (int j = tid; j < NODE; j += 256) {
        float val = wsmat[i * NODE + j] * ninv;
        if (j == i) val -= rowsum * ninv;
        lsbf[(size_t)i * NODE + j] = (bf16_t)val;
    }
}

// ---------------- barrier-counter init ----------------
__global__ void init_kernel(unsigned* cnt) { *cnt = 0u; }

// ---------------- grid-wide barrier (monotonic counter, agent scope) ----------------
__device__ __forceinline__ void grid_barrier(unsigned* cnt, unsigned epoch) {
    __threadfence();          // make prior E stores visible device-wide
    __syncthreads();
    if (threadIdx.x == 0) {
        __hip_atomic_fetch_add(cnt, 1u, __ATOMIC_RELEASE, __HIP_MEMORY_SCOPE_AGENT);
        while (__hip_atomic_load(cnt, __ATOMIC_ACQUIRE, __HIP_MEMORY_SCOPE_AGENT) <
               (unsigned)NWG * epoch) {
            __builtin_amdgcn_s_sleep(1);
        }
    }
    __syncthreads();
    __threadfence();          // acquire remote E stores
}

// ---------------- persistent simulation kernel: 25 WGs x 64 threads ----------------
__global__ void __launch_bounds__(64)
sim_kernel(const float* __restrict__ inp,        // (NODE, HIDDEN, BATCH, 2)
           const float* __restrict__ noise_out,  // (NODE, BATCH)
           const float* __restrict__ hx,         // (NODE, 6)
           const float* __restrict__ gp, const float* __restrict__ gEEp,
           const float* __restrict__ gEIp, const float* __restrict__ gIEp,
           const float* __restrict__ stdinp, const float* __restrict__ stdoutp,
           const bf16_t* __restrict__ lsbf,      // (NODE, NODE) bf16
           bf16_t* __restrict__ Ebf,             // (NODE) bf16, shared E vector
           unsigned* __restrict__ cnt,
           float* __restrict__ out)              // (NODE, BATCH)
{
    __shared__ bf16_t Alds[RPW * NODE];  // 51.2 KB: this WG's 32 rows of l_s
    __shared__ bf16_t Elds[NODE];        // 1.6 KB: current E vector
    __shared__ float  yLds[RPW];         // matvec result for our 32 rows

    const int tid    = threadIdx.x;
    const int wgBase = blockIdx.x * RPW;
    const int lane   = tid & 31;
    const int wave   = tid >> 5;         // tile index: rows [wave*16, wave*16+16)

    // Pin our matrix slice in LDS for the whole simulation (dword copies).
    {
        const unsigned* src = (const unsigned*)(lsbf + (size_t)wgBase * NODE);
        unsigned* dst = (unsigned*)Alds;
        for (int idx = tid; idx < RPW * NODE / 2; idx += 64) dst[idx] = src[idx];
    }

    const float g    = gp[0];
    const float cEE  = 0.001f + fmaxf(gEEp[0], 0.f);
    const float cIE  = 0.001f + fmaxf(gIEp[0], 0.f);
    const float cEI  = 0.001f + fmaxf(gEIp[0], 0.f);
    const float sin_ = 0.02f  + fmaxf(stdinp[0], 0.f);
    const float sout = 0.01f  + fmaxf(stdoutp[0], 0.f);
    const float dt   = 0.005f;
    const float sqdt = 0.07071067811865475f;

    // Per-node ODE state lives in registads of threads 0..31.
    float E = 0.f, I = 0.f, x = 0.f, f = 0.f, v = 0.f, q = 0.f;
    const int n = wgBase + tid;
    if (tid < 32) {
        E = hx[n * 6 + 0]; I = hx[n * 6 + 1]; x = hx[n * 6 + 2];
        f = hx[n * 6 + 3]; v = hx[n * 6 + 4]; q = hx[n * 6 + 5];
        Ebf[n] = (bf16_t)E;
    }

    unsigned epoch = 0;
    grid_barrier(cnt, ++epoch);   // initial E published by all WGs

    // Per-lane WMMA operand addressing (16-bit A 16x32 layout; B's K map mirrors A's).
    const int M     = lane & 15;
    const int baseK = (lane >> 4) * 8;                    // 0 or 8
    const bf16_t* arow = &Alds[(wave * 16 + M) * NODE + baseK];

    for (int b = 0; b < BATCH; ++b) {
        for (int h = 0; h < HIDDEN; ++h) {
            // Broadcast E into LDS (all 64 threads).
            {
                const unsigned* src = (const unsigned*)Ebf;
                unsigned* dst = (unsigned*)Elds;
                for (int idx = tid; idx < NODE / 2; idx += 64) dst[idx] = src[idx];
            }
            __syncthreads();

            // Matvec tile via chained v_wmma_f32_16x16x32_bf16, E broadcast into B cols.
            v8f acc = {0.f, 0.f, 0.f, 0.f, 0.f, 0.f, 0.f, 0.f};
            for (int kb = 0; kb < KBLK; ++kb) {
                v8bf alo = *(const v8bf*)(arow + kb * 32);
                v8bf ahi = *(const v8bf*)(arow + kb * 32 + 16);
                v8bf blo = *(const v8bf*)(&Elds[kb * 32 + baseK]);
                v8bf bhi = *(const v8bf*)(&Elds[kb * 32 + baseK + 16]);
                v16bf a  = __builtin_shufflevector(alo, ahi, 0,1,2,3,4,5,6,7,
                                                   8,9,10,11,12,13,14,15);
                v16bf bm = __builtin_shufflevector(blo, bhi, 0,1,2,3,4,5,6,7,
                                                   8,9,10,11,12,13,14,15);
                acc = __builtin_amdgcn_wmma_f32_16x16x32_bf16(
                          false, a, false, bm, (short)0, acc, false, false);
            }
            // D columns are replicated: lane0 holds rows 0..7, lane16 rows 8..15.
            if (lane == 0) {
                #pragma unroll
                for (int r = 0; r < 8; ++r) yLds[wave * 16 + r] = acc[r];
            } else if (lane == 16) {
                #pragma unroll
                for (int r = 0; r < 8; ++r) yLds[wave * 16 + 8 + r] = acc[r];
            }
            __syncthreads();

            // Elementwise neural-mass + hemodynamic update (threads 0..31 = our nodes).
            if (tid < 32) {
                const float matv = yLds[tid];
                const size_t nidx = (((size_t)n * HIDDEN + h) * BATCH + b) * 2;
                __builtin_prefetch(&inp[nidx + BATCH * 2], 0, 1);  // next euler step's noise
                const float nE = inp[nidx], nI = inp[nidx + 1];

                float IE = tanhf(fmaxf(0.f, 0.2f + cEE * E + g * matv - cIE * I));
                float II = tanhf(fmaxf(0.f, 0.7f * 0.2f + cEI * E - I));
                float uE = 310.f * IE - 125.f;
                float rE = (1e-5f + fabsf(uE)) /
                           (1e-5f * 0.16f + fabsf(1.f - expf(-0.16f * uE)));
                float uI = 615.f * II - 177.f;
                float rI = (1e-5f + fabsf(uI)) /
                           (1e-5f * 0.087f + fabsf(1.f - expf(-0.087f * uI)));
                float ddE = E + dt * (-E / 100.f + 0.000641f * (1.f - E) * rE) + sqdt * nE * sin_;
                float ddI = I + dt * (-I / 10.f + 0.001f * rI) + sqdt * nI * sin_;
                E = tanhf(1e-5f + fmaxf(0.f, ddE));
                I = tanhf(1e-5f + fmaxf(0.f, ddI));

                float xn = x + dt * (E - x / 0.65f - (f - 1.f) / 0.41f);
                float fn = f + dt * x;
                float vp = powf(v, 3.125f);                       // v^(1/alpha)
                float vn = v + dt * (f - vp) / 0.98f;
                float qn = q + dt * (f * (1.f - powf(0.66f, 1.f / f)) / 0.34f
                                     - q * vp / v) / 0.98f;
                x = xn; f = fn; v = vn; q = qn;

                Ebf[n] = (bf16_t)E;

                if (h == HIDDEN - 1) {
                    float nb = noise_out[n * BATCH + b];
                    float bold = sout * nb + (100.f * 0.02f / 0.34f) *
                                 (2.38f * (1.f - q) + 2.0f * (1.f - q / v) + 0.48f * (1.f - v));
                    out[n * BATCH + b] = bold;
                }
            }
            grid_barrier(cnt, ++epoch);   // publish E, sync all 25 WGs
        }
    }
}

// ---------------- host side ----------------
extern "C" void kernel_launch(void* const* d_in, const int* in_sizes, int n_in,
                              void* d_out, int out_size, void* d_ws, size_t ws_size,
                              hipStream_t stream) {
    (void)in_sizes; (void)n_in; (void)out_size; (void)ws_size;
    const float* inp       = (const float*)d_in[0];
    const float* noise_out = (const float*)d_in[1];
    const float* hx        = (const float*)d_in[2];
    const float* sc        = (const float*)d_in[3];
    const float* wbb       = (const float*)d_in[4];
    const float* g         = (const float*)d_in[5];
    const float* gEE       = (const float*)d_in[6];
    const float* gEI       = (const float*)d_in[7];
    const float* gIE       = (const float*)d_in[8];
    const float* std_in    = (const float*)d_in[9];
    const float* std_out   = (const float*)d_in[10];
    float* out = (float*)d_out;

    // workspace layout (16B aligned offsets)
    char* wsb = (char*)d_ws;
    float*    wsmat    = (float*)(wsb);                                  // 2,560,000 B
    float*    partials = (float*)(wsb + 2560000);                        //     3,200 B
    bf16_t*   lsbf     = (bf16_t*)(wsb + 2560000 + 3200);                // 1,280,000 B
    bf16_t*   Ebf      = (bf16_t*)(wsb + 2560000 + 3200 + 1280000);      //     1,600 B
    unsigned* cnt      = (unsigned*)(wsb + 2560000 + 3200 + 1280000 + 1600);

    prep1_kernel<<<NODE, 256, 0, stream>>>(wbb, sc, wsmat, partials);
    prep2_kernel<<<NODE, 256, 0, stream>>>(wsmat, partials, lsbf);
    init_kernel<<<1, 1, 0, stream>>>(cnt);
    sim_kernel<<<NWG, 64, 0, stream>>>(inp, noise_out, hx,
                                       g, gEE, gEI, gIE, std_in, std_out,
                                       lsbf, Ebf, cnt, out);
}
